// MOSGEN_27797028339917
// MI455X (gfx1250) — matrix operand order repaired and checked by reading, not compiled, not run on hardware
//
#include <hip/hip_runtime.h>
#include <hip/hip_bf16.h>
#include <math.h>

// ---------------------------------------------------------------------------
// Sizes (fixed by the reference problem)
// ---------------------------------------------------------------------------
#define NN 100000      // nodes
#define NE 1600000     // edges
#define NG 1024        // graphs
#define DN 128         // node feature dim
#define DE 32          // edge feature dim
#define C1 64          // layer-1 channels
#define C2 16          // layer-2 channels

typedef __attribute__((ext_vector_type(16))) __bf16 v16bf;
typedef __attribute__((ext_vector_type(8)))  float  v8f;

// ---------------------------------------------------------------------------
// Helpers
// ---------------------------------------------------------------------------
__device__ __forceinline__ unsigned f2ord(float f) {
    unsigned u = __float_as_uint(f);
    return (u & 0x80000000u) ? ~u : (u | 0x80000000u);
}
__device__ __forceinline__ float ord2f(unsigned u) {
    u = (u & 0x80000000u) ? (u & 0x7fffffffu) : ~u;
    return __uint_as_float(u);
}
__device__ __forceinline__ void fatomic_add(float* p, float v) {
    __hip_atomic_fetch_add(p, v, __ATOMIC_RELAXED, __HIP_MEMORY_SCOPE_AGENT);
}

__global__ void fill_f32(float* __restrict__ p, float v, int n) {
    int i = blockIdx.x * blockDim.x + threadIdx.x;
    if (i < n) p[i] = v;
}
__global__ void fill_u32(unsigned* __restrict__ p, unsigned v, int n) {
    int i = blockIdx.x * blockDim.x + threadIdx.x;
    if (i < n) p[i] = v;
}
__global__ void relu_k(float* __restrict__ p, int n) {
    int i = blockIdx.x * blockDim.x + threadIdx.x;
    if (i < n) p[i] = fmaxf(p[i], 0.0f);
}

// ---------------------------------------------------------------------------
// WMMA GEMM:  Y[nrows x cout] = bf16(X[nrows x CIN]) @ bf16(W[CIN x cout]) + b
//
// One wave owns a 16-wide column tile and grid-strides over 16-row tiles.
// B fragments (per column tile) are loaded ONCE into registers and reused
// across all row tiles. A fragments load as 4 x float4 (global_load_b128) per
// k-step: for lane-half h, v16bf elements 0..7 are K = h*8+0..7 and elements
// 8..15 are K = 16+h*8+0..7 -> two contiguous 32B chunks.
// Ragged last tile: row index clamped (A row m only feeds D row m, which is
// simply not stored), so no per-element predication.
// ---------------------------------------------------------------------------
template <int CIN>
__global__ void wmma_gemm(const float* __restrict__ X,
                          const float* __restrict__ W,
                          const float* __restrict__ bias,
                          float* __restrict__ Y,
                          int nrows, int cout)
{
    constexpr int KSTEPS = CIN / 32;
    const int lane = threadIdx.x & 31;
    const int wave = threadIdx.x >> 5;
    const int m    = lane & 15;
    const int half = lane >> 4;
    const int col  = blockIdx.y * 16 + m;      // N index for B and C/D

    // ---- B fragments: element j holds W[ks*32 + half*8 + j + (j & 8)][col] ----
    v16bf bfrag[KSTEPS];
#pragma unroll
    for (int ks = 0; ks < KSTEPS; ++ks) {
        const int kbase = ks * 32 + half * 8;
#pragma unroll
        for (int j = 0; j < 16; ++j)
            bfrag[ks][j] = (__bf16)W[(size_t)(kbase + j + (j & 8)) * cout + col];
    }
    const float bv = bias ? bias[col] : 0.0f;

    const int wpb        = blockDim.x >> 5;
    const int tilesTotal = (nrows + 15) >> 4;
    const int stride     = gridDim.x * wpb;

    for (int tile = blockIdx.x * wpb + wave; tile < tilesTotal; tile += stride) {
        const int m0   = tile * 16;
        const int arow = min(m0 + m, nrows - 1);      // clamp; OOB rows never stored
        const float* xp = X + (size_t)arow * CIN;

        v8f acc;
#pragma unroll
        for (int r = 0; r < 8; ++r) acc[r] = bv;

#pragma unroll
        for (int ks = 0; ks < KSTEPS; ++ks) {
            const float4 c0 = *(const float4*)(xp + ks * 32 + half * 8);
            const float4 c1 = *(const float4*)(xp + ks * 32 + half * 8 + 4);
            const float4 c2 = *(const float4*)(xp + ks * 32 + 16 + half * 8);
            const float4 c3 = *(const float4*)(xp + ks * 32 + 16 + half * 8 + 4);
            v16bf a;
            a[0]  = (__bf16)c0.x; a[1]  = (__bf16)c0.y;
            a[2]  = (__bf16)c0.z; a[3]  = (__bf16)c0.w;
            a[4]  = (__bf16)c1.x; a[5]  = (__bf16)c1.y;
            a[6]  = (__bf16)c1.z; a[7]  = (__bf16)c1.w;
            a[8]  = (__bf16)c2.x; a[9]  = (__bf16)c2.y;
            a[10] = (__bf16)c2.z; a[11] = (__bf16)c2.w;
            a[12] = (__bf16)c3.x; a[13] = (__bf16)c3.y;
            a[14] = (__bf16)c3.z; a[15] = (__bf16)c3.w;
            acc = __builtin_amdgcn_wmma_f32_16x16x32_bf16(
                false, a, false, bfrag[ks], (short)0, acc, false, false);
        }

        if (m0 + 16 <= nrows) {                        // fast path: no guards
#pragma unroll
            for (int r = 0; r < 8; ++r)
                Y[(size_t)(m0 + r + half * 8) * cout + col] = acc[r];
        } else {                                       // single ragged tile
#pragma unroll
            for (int r = 0; r < 8; ++r) {
                const int row = m0 + r + half * 8;
                if (row < nrows) Y[(size_t)row * cout + col] = acc[r];
            }
        }
    }
}

// ---------------------------------------------------------------------------
// Edge pass 1: score[e] = dot(q[dst], k[src] + edge_attr@We) * scale,
//              segmax[dst] = max(...)  (order-preserving uint atomicMax)
// We (32 x C) staged in LDS; e recomputed on the fly (cheaper than
// materializing a 410 MB E x C tensor: 205 MB read vs 1.2 GB of traffic).
// ---------------------------------------------------------------------------
__global__ void edge_score(const float* __restrict__ Q, const float* __restrict__ K,
                           const int* __restrict__ src, const int* __restrict__ dst,
                           const float* __restrict__ EA, const float* __restrict__ We,
                           float* __restrict__ score, unsigned* __restrict__ segmax,
                           int E, int C, float scale)
{
    __shared__ float sWe[DE * C1];           // max 32*64 = 8 KB
    const int nW = DE * C;
    for (int i = threadIdx.x; i < nW; i += blockDim.x) sWe[i] = We[i];
    __syncthreads();

    const int e = blockIdx.x * blockDim.x + threadIdx.x;
    if (e >= E) return;
    const int s = src[e], d = dst[e];

    float ea[DE];
    const float* eap = EA + (size_t)e * DE;
#pragma unroll
    for (int k = 0; k < DE; ++k) ea[k] = eap[k];

    const float* qp = Q + (size_t)d * C;
    const float* kp = K + (size_t)s * C;
    float acc = 0.0f;
    for (int c = 0; c < C; ++c) {
        float ec = 0.0f;
#pragma unroll
        for (int k = 0; k < DE; ++k) ec = fmaf(ea[k], sWe[k * C + c], ec);
        acc = fmaf(qp[c], kp[c] + ec, acc);
    }
    acc *= scale;
    score[e] = acc;
    atomicMax(&segmax[d], f2ord(acc));
}

// ---------------------------------------------------------------------------
// Edge pass 2: a = exp(score - max[dst]); den[dst] += a
// ---------------------------------------------------------------------------
__global__ void edge_exp(const int* __restrict__ dst, float* __restrict__ score,
                         const unsigned* __restrict__ segmax, float* __restrict__ den,
                         int E)
{
    const int e = blockIdx.x * blockDim.x + threadIdx.x;
    if (e >= E) return;
    const int d = dst[e];
    const float a = __expf(score[e] - ord2f(segmax[d]));
    score[e] = a;
    fatomic_add(&den[d], a);
}

// ---------------------------------------------------------------------------
// Edge pass 3: H[dst] += (v[src] + edge_attr@We) * (a / (den[dst]+1e-16))
// ---------------------------------------------------------------------------
__global__ void edge_agg(const float* __restrict__ V,
                         const int* __restrict__ src, const int* __restrict__ dst,
                         const float* __restrict__ EA, const float* __restrict__ We,
                         const float* __restrict__ score, const float* __restrict__ den,
                         float* __restrict__ H, int E, int C)
{
    __shared__ float sWe[DE * C1];
    const int nW = DE * C;
    for (int i = threadIdx.x; i < nW; i += blockDim.x) sWe[i] = We[i];
    __syncthreads();

    const int e = blockIdx.x * blockDim.x + threadIdx.x;
    if (e >= E) return;
    const int s = src[e], d = dst[e];
    const float alpha = score[e] / (den[d] + 1e-16f);

    float ea[DE];
    const float* eap = EA + (size_t)e * DE;
#pragma unroll
    for (int k = 0; k < DE; ++k) ea[k] = eap[k];

    const float* vp = V + (size_t)s * C;
    float* hp = H + (size_t)d * C;
    for (int c = 0; c < C; ++c) {
        float ec = 0.0f;
#pragma unroll
        for (int k = 0; k < DE; ++k) ec = fmaf(ea[k], sWe[k * C + c], ec);
        fatomic_add(&hp[c], (vp[c] + ec) * alpha);
    }
}

// ---------------------------------------------------------------------------
// Mean-pool (fuses the final ReLU): pool[batch[i]] += relu(H2[i]); cnt[g] += 1
// ---------------------------------------------------------------------------
__global__ void pool_k(const float* __restrict__ H2, const int* __restrict__ batch,
                       float* __restrict__ pool, float* __restrict__ cnt, int n)
{
    const int i = blockIdx.x * blockDim.x + threadIdx.x;
    if (i >= n) return;
    const int g = batch[i];
    const float* hp = H2 + (size_t)i * C2;
    float* pp = pool + (size_t)g * C2;
#pragma unroll
    for (int c = 0; c < C2; ++c) fatomic_add(&pp[c], fmaxf(hp[c], 0.0f));
    fatomic_add(&cnt[g], 1.0f);
}

// ---------------------------------------------------------------------------
// Final: out[g] = sigmoid((pool[g]/max(cnt,1)) @ Wf + bf)
// ---------------------------------------------------------------------------
__global__ void final_k(const float* __restrict__ pool, const float* __restrict__ cnt,
                        const float* __restrict__ Wf, const float* __restrict__ bfp,
                        float* __restrict__ out, int G)
{
    const int g = blockIdx.x * blockDim.x + threadIdx.x;
    if (g >= G) return;
    const float inv = 1.0f / fmaxf(cnt[g], 1.0f);
    float s = bfp[0];
#pragma unroll
    for (int c = 0; c < C2; ++c) s = fmaf(pool[g * C2 + c] * inv, Wf[c], s);
    out[g] = 1.0f / (1.0f + __expf(-s));
}

// ---------------------------------------------------------------------------
// Launch
// ---------------------------------------------------------------------------
static inline int cdiv(int a, int b) { return (a + b - 1) / b; }

extern "C" void kernel_launch(void* const* d_in, const int* in_sizes, int n_in,
                              void* d_out, int out_size, void* d_ws, size_t ws_size,
                              hipStream_t stream)
{
    const float* x   = (const float*)d_in[0];
    const float* ea  = (const float*)d_in[1];
    const float* Wq1 = (const float*)d_in[2];  const float* bq1 = (const float*)d_in[3];
    const float* Wk1 = (const float*)d_in[4];  const float* bk1 = (const float*)d_in[5];
    const float* Wv1 = (const float*)d_in[6];  const float* bv1 = (const float*)d_in[7];
    const float* We1 = (const float*)d_in[8];
    const float* Ws1 = (const float*)d_in[9];  const float* bs1 = (const float*)d_in[10];
    const float* Wq2 = (const float*)d_in[11]; const float* bq2 = (const float*)d_in[12];
    const float* Wk2 = (const float*)d_in[13]; const float* bk2 = (const float*)d_in[14];
    const float* Wv2 = (const float*)d_in[15]; const float* bv2 = (const float*)d_in[16];
    const float* We2 = (const float*)d_in[17];
    const float* Ws2 = (const float*)d_in[18]; const float* bs2 = (const float*)d_in[19];
    const float* Wf  = (const float*)d_in[20]; const float* bf_ = (const float*)d_in[21];
    const int*   ei    = (const int*)d_in[22];
    const int*   batch = (const int*)d_in[23];
    const int* srcI = ei;            // edge_index[0, :]
    const int* dstI = ei + NE;       // edge_index[1, :]

    // -------- workspace carve-out (~135 MB) --------
    char* w = (char*)d_ws;
    auto carve = [&](size_t bytes) -> void* {
        void* p = (void*)w;
        w += (bytes + 255) & ~(size_t)255;
        return p;
    };
    float*    Q1  = (float*)carve((size_t)NN * C1 * 4);
    float*    K1  = (float*)carve((size_t)NN * C1 * 4);
    float*    V1  = (float*)carve((size_t)NN * C1 * 4);
    float*    H   = (float*)carve((size_t)NN * C1 * 4);   // skip + aggregate, layer 1
    float*    Q2  = (float*)carve((size_t)NN * C2 * 4);
    float*    K2b = (float*)carve((size_t)NN * C2 * 4);
    float*    V2  = (float*)carve((size_t)NN * C2 * 4);
    float*    H2  = (float*)carve((size_t)NN * C2 * 4);   // skip + aggregate, layer 2
    float*    sc  = (float*)carve((size_t)NE * 4);        // score, then exp(score-max)
    unsigned* smx = (unsigned*)carve((size_t)NN * 4);
    float*    den = (float*)carve((size_t)NN * 4);
    float*    pl  = (float*)carve((size_t)NG * C2 * 4);
    float*    cnt = (float*)carve((size_t)NG * 4);
    float*    out = (float*)d_out;

    const unsigned NEG_INF_ORD = 0x007FFFFFu;  // f2ord(-inf)
    const int TB = 256;
    const int nodeBlocks = cdiv(NN, TB);
    const int edgeBlocks = cdiv(NE, TB);

    // GEMM launch: 8 waves/block; each wave grid-strides over ~4 row tiles so
    // the register-resident B fragments are reused.
    const int tiles   = cdiv(NN, 16);
    const int gemmGrx = cdiv(tiles, 8 * 4);

    // ---------------- Layer 1 ----------------
    fill_u32<<<nodeBlocks, TB, 0, stream>>>(smx, NEG_INF_ORD, NN);
    fill_f32<<<nodeBlocks, TB, 0, stream>>>(den, 0.0f, NN);

    {
        dim3 g(gemmGrx, C1 / 16);
        wmma_gemm<DN><<<g, 256, 0, stream>>>(x, Wq1, bq1, Q1, NN, C1);
        wmma_gemm<DN><<<g, 256, 0, stream>>>(x, Wk1, bk1, K1, NN, C1);
        wmma_gemm<DN><<<g, 256, 0, stream>>>(x, Wv1, bv1, V1, NN, C1);
        wmma_gemm<DN><<<g, 256, 0, stream>>>(x, Ws1, bs1, H,  NN, C1);  // skip init
    }

    edge_score<<<edgeBlocks, TB, 0, stream>>>(Q1, K1, srcI, dstI, ea, We1,
                                              sc, smx, NE, C1, 0.125f /*1/sqrt(64)*/);
    edge_exp<<<edgeBlocks, TB, 0, stream>>>(dstI, sc, smx, den, NE);
    edge_agg<<<edgeBlocks, TB, 0, stream>>>(V1, srcI, dstI, ea, We1, sc, den, H, NE, C1);
    relu_k<<<cdiv(NN * C1, TB), TB, 0, stream>>>(H, NN * C1);

    // ---------------- Layer 2 ----------------
    fill_u32<<<nodeBlocks, TB, 0, stream>>>(smx, NEG_INF_ORD, NN);
    fill_f32<<<nodeBlocks, TB, 0, stream>>>(den, 0.0f, NN);

    {
        dim3 g(gemmGrx, C2 / 16);
        wmma_gemm<C1><<<g, 256, 0, stream>>>(H, Wq2, bq2, Q2,  NN, C2);
        wmma_gemm<C1><<<g, 256, 0, stream>>>(H, Wk2, bk2, K2b, NN, C2);
        wmma_gemm<C1><<<g, 256, 0, stream>>>(H, Wv2, bv2, V2,  NN, C2);
        wmma_gemm<C1><<<g, 256, 0, stream>>>(H, Ws2, bs2, H2,  NN, C2); // skip init
    }

    edge_score<<<edgeBlocks, TB, 0, stream>>>(Q2, K2b, srcI, dstI, ea, We2,
                                              sc, smx, NE, C2, 0.25f /*1/sqrt(16)*/);
    edge_exp<<<edgeBlocks, TB, 0, stream>>>(dstI, sc, smx, den, NE);
    edge_agg<<<edgeBlocks, TB, 0, stream>>>(V2, srcI, dstI, ea, We2, sc, den, H2, NE, C2);

    // ---------------- Pool + head ----------------
    fill_f32<<<cdiv(NG * C2, TB), TB, 0, stream>>>(pl, 0.0f, NG * C2);
    fill_f32<<<cdiv(NG, TB), TB, 0, stream>>>(cnt, 0.0f, NG);
    pool_k<<<nodeBlocks, TB, 0, stream>>>(H2, batch, pl, cnt, NN);
    final_k<<<cdiv(NG, TB), TB, 0, stream>>>(pl, cnt, Wf, bf_, out, NG);
}
